// MHSA3D_58076547777275
// MI455X (gfx1250) — compile-verified, matrix operand order checked by reading, not run on hardware
//
#include <hip/hip_runtime.h>

#define B 4
#define L 2052
#define C 1024
#define H 16
#define D 64
#define NREG 4
#define LSP (L - NREG)          // 2048
#define M_ROWS (B * L)          // 8208 (multiple of 16)
#define NQKV (3 * C)            // 3072
#define MTILES ((M_ROWS + 15) / 16)   // 513
#define QTILES ((L + 15) / 16)        // 129

typedef __attribute__((ext_vector_type(16))) __bf16 v16bf;
typedef __attribute__((ext_vector_type(8)))  __bf16 v8bf;
typedef __attribute__((ext_vector_type(8)))  float  v8f;
typedef __attribute__((ext_vector_type(4)))  unsigned int v4u;
typedef __attribute__((ext_vector_type(8)))  int v8i;
typedef __attribute__((ext_vector_type(4)))  int v4i;

__device__ __forceinline__ v16bf join16(v8bf lo, v8bf hi) {
    v16bf r;
#pragma unroll
    for (int i = 0; i < 8; ++i) { r[i] = lo[i]; r[i + 8] = hi[i]; }
    return r;
}

__device__ __forceinline__ v8f zero8() {
    v8f z;
#pragma unroll
    for (int i = 0; i < 8; ++i) z[i] = 0.0f;
    return z;
}

// ---------------------------------------------------------------- fp32 -> bf16
__global__ void MHSA_cvt_bf16(const float* __restrict__ in,
                              __bf16* __restrict__ out, int n) {
    int i = blockIdx.x * blockDim.x + threadIdx.x;
    if (i < n) out[i] = (__bf16)in[i];
}

// ---------------------------------------------------------------- QKV GEMM
// X (8208 x 1024) bf16 @ Wqkv (1024 x 3072) bf16 + bqkv -> scatter q/k/v bf16 (B,H,L,D)
__global__ __launch_bounds__(128)
void MHSA_gemm_qkv(const __bf16* __restrict__ X, const __bf16* __restrict__ W,
                   const float* __restrict__ bias,
                   __bf16* __restrict__ qb, __bf16* __restrict__ kb,
                   __bf16* __restrict__ vb) {
    __shared__ __bf16 ldsW[64 * 32];          // transposed tile: [n][k]
    const int wid   = threadIdx.x >> 5;
    const int lane  = threadIdx.x & 31;
    const int nl    = lane & 15;
    const int g     = lane >> 4;
    const int mtile = blockIdx.x * 4 + wid;
    const int n0    = blockIdx.y * 64;

    int arow = mtile * 16 + nl;
    if (arow >= M_ROWS) arow = M_ROWS - 1;
    const __bf16* Xrow = X + (size_t)arow * C;

    v8f acc[4];
#pragma unroll
    for (int t = 0; t < 4; ++t) acc[t] = zero8();

    for (int k0 = 0; k0 < C; k0 += 32) {
        // cooperative transposed staging of W tile (32 x 64)
        {
            const int kk = threadIdx.x >> 2;
            const int nb = (threadIdx.x & 3) * 16;
            const __bf16* src = W + (size_t)(k0 + kk) * NQKV + n0 + nb;
            v8bf a = *(const v8bf*)(src);
            v8bf b = *(const v8bf*)(src + 8);
            if (k0 + 32 < C)
                __builtin_prefetch((const void*)(src + 32 * (size_t)NQKV), 0, 1);
#pragma unroll
            for (int i = 0; i < 8; ++i) {
                ldsW[(nb + i) * 32 + kk]     = a[i];
                ldsW[(nb + 8 + i) * 32 + kk] = b[i];
            }
        }
        __syncthreads();
        // A fragment: two contiguous 16B chunks per lane
        v8bf alo = *(const v8bf*)(Xrow + k0 + g * 8);
        v8bf ahi = *(const v8bf*)(Xrow + k0 + 16 + g * 8);
        v16bf afrag = join16(alo, ahi);
#pragma unroll
        for (int t = 0; t < 4; ++t) {
            const __bf16* wp = &ldsW[(t * 16 + nl) * 32];
            v8bf blo = *(const v8bf*)(wp + g * 8);
            v8bf bhi = *(const v8bf*)(wp + 16 + g * 8);
            v16bf bfrag = join16(blo, bhi);
            acc[t] = __builtin_amdgcn_wmma_f32_16x16x32_bf16(
                false, afrag, false, bfrag, (short)0, acc[t], false, false);
        }
        __syncthreads();
    }

    if (mtile < MTILES) {
#pragma unroll
        for (int t = 0; t < 4; ++t) {
            const int c     = n0 + t * 16 + nl;
            const int which = c >> 10;
            const int rem   = c & 1023;
            const int h     = rem >> 6;
            const int d     = rem & 63;
            __bf16* dst = (which == 0) ? qb : ((which == 1) ? kb : vb);
            const float bv = bias[c];
#pragma unroll
            for (int r = 0; r < 8; ++r) {
                const int orow = mtile * 16 + r + 8 * g;
                const int bb = orow / L, ll = orow % L;
                dst[(((size_t)bb * H + h) * L + ll) * D + d] =
                    (__bf16)(acc[t][r] + bv);
            }
        }
    }
}

// ---------------------------------------------------------------- RoPE (in place on bf16 q/k)
__global__ void MHSA_rope(__bf16* __restrict__ qb, __bf16* __restrict__ kb,
                          const float* __restrict__ rope) {
    const int total = B * H * LSP * 32;
    int idx = blockIdx.x * blockDim.x + threadIdx.x;
    if (idx >= total) return;
    const int d    = idx & 31;
    const int rest = idx >> 5;
    const int lsp  = rest % LSP;
    const int bh   = rest / LSP;
    __bf16* buf = blockIdx.y ? kb : qb;
    const size_t base = ((size_t)bh * L + (lsp + NREG)) * D;
    const float x1 = (float)buf[base + d];
    const float x2 = (float)buf[base + d + 32];
    const float* cs = rope + (size_t)lsp * D;
    const float* sn = rope + (size_t)LSP * D + (size_t)lsp * D;
    buf[base + d]      = (__bf16)(x1 * cs[d]      - x2 * sn[d]);
    buf[base + d + 32] = (__bf16)(x2 * cs[d + 32] + x1 * sn[d + 32]);
}

// ---------------------------------------------------------------- flash attention
// Block = 4 waves sharing one (b,h); wave w owns q-tile blockIdx.x*4+w.
// V chunks (32 keys x 64 dims, bf16) are DMA'd once per block into LDS by the
// Tensor Data Mover (wave 0), zero-filling keys >= L via tensor_dim1.
__global__ __launch_bounds__(128)
void MHSA_attn(const __bf16* __restrict__ qb, const __bf16* __restrict__ kb,
               const __bf16* __restrict__ vb, float* __restrict__ out) {
    __shared__ __bf16 ldsV[32 * 64];          // V chunk row-major [kc][d], block-shared
    __shared__ __bf16 ldsP[4][16 * 32];       // per-wave probs staging
    const int wid  = threadIdx.x >> 5;
    const int lane = threadIdx.x & 31;
    const int nl   = lane & 15;
    const int g    = lane >> 4;
    int qtile = blockIdx.x * 4 + wid;
    const bool qvalid = qtile < QTILES;
    if (!qvalid) qtile = QTILES - 1;          // keep wave alive for barriers
    const int bh = blockIdx.y;
    const int b_ = bh >> 4;
    const int h_ = bh & 15;
    const size_t bhBase = (size_t)bh * L * D;
    const unsigned ldsVaddr = (unsigned)(size_t)(void*)&ldsV[0];

    // preload Q A-fragments (16 rows x 64 dims = 2 fragments)
    int qrow = qtile * 16 + nl;
    if (qrow >= L) qrow = L - 1;
    const __bf16* qp = qb + bhBase + (size_t)qrow * D;
    v16bf qa[2];
#pragma unroll
    for (int ks = 0; ks < 2; ++ks) {
        v8bf lo = *(const v8bf*)(qp + ks * 32 + g * 8);
        v8bf hi = *(const v8bf*)(qp + ks * 32 + 16 + g * 8);
        qa[ks] = join16(lo, hi);
    }

    v8f acc[4];
    float mrow[8], lrow[8];
#pragma unroll
    for (int t = 0; t < 4; ++t) acc[t] = zero8();
#pragma unroll
    for (int r = 0; r < 8; ++r) { mrow[r] = -1.0e30f; lrow[r] = 0.0f; }

    const float scale = 0.125f;               // 1/sqrt(64)
    __bf16* myP = ldsP[wid];

    for (int kc0 = 0; kc0 < L; kc0 += 32) {
        __syncthreads();                      // prior iteration done reading ldsV
        if (wid == 0) {
            // ---- TDM: DMA V[kc0 : kc0+32, 0:64] (bf16) into ldsV, row-major.
            const unsigned long long ga =
                (unsigned long long)(size_t)(vb + bhBase + (size_t)kc0 * D);
            const unsigned rem = (unsigned)(L - kc0);   // rows in-bounds; rest zero-fill
            v4u g0;
            g0[0] = 1u;                                  // count=1, user mode
            g0[1] = ldsVaddr;                            // lds_addr
            g0[2] = (unsigned)(ga & 0xffffffffu);        // global_addr[31:0]
            g0[3] = (unsigned)((ga >> 32) & 0x1ffffffu)  // global_addr[56:32]
                    | (2u << 30);                        // type = 2 (image)
            v8i g1;
            g1[0] = (int)(1u << 16);                     // data_size = 2 bytes
            g1[1] = (int)(64u << 16);                    // tensor_dim0 = 64 (lo16)
            g1[2] = (int)((rem & 0xffffu) << 16);        // dim0 hi16 | tensor_dim1 lo16
            g1[3] = (int)(((rem >> 16) & 0xffffu) | (64u << 16)); // dim1 hi | tile_dim0=64
            g1[4] = 32;                                  // tile_dim1=32, tile_dim2=0
            g1[5] = 64;                                  // tensor_dim0_stride = 64 (lo32)
            g1[6] = (int)(64u << 16);                    // stride0 hi16 | stride1 lo16 = 64
            g1[7] = 0;                                   // stride1 hi32
            v4i gz; gz[0] = gz[1] = gz[2] = gz[3] = 0;
#if defined(__clang_major__) && (__clang_major__ >= 23)
            v8i gz8;
#pragma unroll
            for (int i = 0; i < 8; ++i) gz8[i] = 0;
            __builtin_amdgcn_tensor_load_to_lds(g0, g1, gz, gz, gz8, 0);
#else
            __builtin_amdgcn_tensor_load_to_lds(g0, g1, gz, gz, 0);
#endif
            __builtin_amdgcn_s_wait_tensorcnt(0);
        }
        __syncthreads();                      // ldsV ready for all waves

        // ---- S = Q K^T for two 16-key tiles
        v8f s[2];
#pragma unroll
        for (int t = 0; t < 2; ++t) {
            const int key = kc0 + t * 16 + nl;
            const int ck  = (key < L) ? key : (L - 1);
            const __bf16* kp = kb + bhBase + (size_t)ck * D;
            v8f st = zero8();
#pragma unroll
            for (int ks = 0; ks < 2; ++ks) {
                v8bf lo = *(const v8bf*)(kp + ks * 32 + g * 8);
                v8bf hi = *(const v8bf*)(kp + ks * 32 + 16 + g * 8);
                v16bf kf = join16(lo, hi);
                st = __builtin_amdgcn_wmma_f32_16x16x32_bf16(
                    false, qa[ks], false, kf, (short)0, st, false, false);
            }
            const bool valid = key < L;
#pragma unroll
            for (int r = 0; r < 8; ++r)
                st[r] = valid ? st[r] * scale : -1.0e30f;
            s[t] = st;
        }
        // ---- online softmax (rows m = r + 8*g live across 16 lanes)
#pragma unroll
        for (int r = 0; r < 8; ++r) {
            float mx = fmaxf(s[0][r], s[1][r]);
#pragma unroll
            for (int off = 1; off < 16; off <<= 1)
                mx = fmaxf(mx, __shfl_xor(mx, off, 32));
            const float mnew  = fmaxf(mrow[r], mx);
            const float alpha = __expf(mrow[r] - mnew);
            mrow[r] = mnew;
            const float p0 = __expf(s[0][r] - mnew);
            const float p1 = __expf(s[1][r] - mnew);
            float rs = p0 + p1;
#pragma unroll
            for (int off = 1; off < 16; off <<= 1)
                rs += __shfl_xor(rs, off, 32);
            lrow[r] = lrow[r] * alpha + rs;
#pragma unroll
            for (int t = 0; t < 4; ++t) acc[t][r] *= alpha;
            const int m = r + 8 * g;
            myP[m * 32 + nl]      = (__bf16)p0;
            myP[m * 32 + 16 + nl] = (__bf16)p1;
        }
        asm volatile("s_wait_dscnt 0" ::: "memory");
        // ---- PV: A = probs (16x32), B = V chunk (32x16 per d-tile) from ldsV
        {
            v8bf plo = *(const v8bf*)(&myP[nl * 32 + g * 8]);
            v8bf phi = *(const v8bf*)(&myP[nl * 32 + 16 + g * 8]);
            v16bf pf = join16(plo, phi);
#pragma unroll
            for (int dt = 0; dt < 4; ++dt) {
                const int d = dt * 16 + nl;
                v16bf vf;
#pragma unroll
                for (int e = 0; e < 16; ++e) {
                    const int kc = (e < 8) ? (g * 8 + e) : (16 + g * 8 + (e - 8));
                    vf[e] = ldsV[kc * 64 + d];
                }
                acc[dt] = __builtin_amdgcn_wmma_f32_16x16x32_bf16(
                    false, pf, false, vf, (short)0, acc[dt], false, false);
            }
        }
        asm volatile("s_wait_dscnt 0" ::: "memory");
    }
    // write O = acc / l
    if (qvalid) {
#pragma unroll
        for (int dt = 0; dt < 4; ++dt) {
#pragma unroll
            for (int r = 0; r < 8; ++r) {
                const int m = r + 8 * g;
                const int l = qtile * 16 + m;
                if (l < L)
                    out[((size_t)b_ * L + l) * C + h_ * D + dt * 16 + nl] =
                        acc[dt][r] / lrow[r];
            }
        }
    }
}

// ---------------------------------------------------------------- LayerNorm -> bf16
__global__ __launch_bounds__(256)
void MHSA_ln(const float* __restrict__ in, const float* __restrict__ gamma,
             const float* __restrict__ beta, __bf16* __restrict__ out) {
    __shared__ float red[16];
    const int row = blockIdx.x;
    const float* x = in + (size_t)row * C;
    float s = 0.0f, ss = 0.0f;
#pragma unroll
    for (int i = 0; i < 4; ++i) {
        const float v = x[threadIdx.x + i * 256];
        s += v; ss += v * v;
    }
#pragma unroll
    for (int off = 1; off < 32; off <<= 1) {
        s  += __shfl_xor(s, off, 32);
        ss += __shfl_xor(ss, off, 32);
    }
    const int wid = threadIdx.x >> 5, lane = threadIdx.x & 31;
    if (lane == 0) { red[wid] = s; red[8 + wid] = ss; }
    __syncthreads();
    if (threadIdx.x == 0) {
        float ts = 0.0f, tss = 0.0f;
        for (int i = 0; i < 8; ++i) { ts += red[i]; tss += red[8 + i]; }
        red[0] = ts; red[8] = tss;
    }
    __syncthreads();
    const float mu   = red[0] / (float)C;
    const float var  = red[8] / (float)C - mu * mu;
    const float rstd = rsqrtf(var + 1e-5f);
    __bf16* o = out + (size_t)row * C;
#pragma unroll
    for (int i = 0; i < 4; ++i) {
        const int c = threadIdx.x + i * 256;
        o[c] = (__bf16)((x[c] - mu) * rstd * gamma[c] + beta[c]);
    }
}

// ---------------------------------------------------------------- Proj GEMM -> fp32 out
__global__ __launch_bounds__(128)
void MHSA_gemm_proj(const __bf16* __restrict__ X, const __bf16* __restrict__ W,
                    const float* __restrict__ bias, float* __restrict__ out) {
    __shared__ __bf16 ldsW[64 * 32];
    const int wid   = threadIdx.x >> 5;
    const int lane  = threadIdx.x & 31;
    const int nl    = lane & 15;
    const int g     = lane >> 4;
    const int mtile = blockIdx.x * 4 + wid;
    const int n0    = blockIdx.y * 64;

    int arow = mtile * 16 + nl;
    if (arow >= M_ROWS) arow = M_ROWS - 1;
    const __bf16* Xrow = X + (size_t)arow * C;

    v8f acc[4];
#pragma unroll
    for (int t = 0; t < 4; ++t) acc[t] = zero8();

    for (int k0 = 0; k0 < C; k0 += 32) {
        {
            const int kk = threadIdx.x >> 2;
            const int nb = (threadIdx.x & 3) * 16;
            const __bf16* src = W + (size_t)(k0 + kk) * C + n0 + nb;
            v8bf a = *(const v8bf*)(src);
            v8bf b = *(const v8bf*)(src + 8);
            if (k0 + 32 < C)
                __builtin_prefetch((const void*)(src + 32 * (size_t)C), 0, 1);
#pragma unroll
            for (int i = 0; i < 8; ++i) {
                ldsW[(nb + i) * 32 + kk]     = a[i];
                ldsW[(nb + 8 + i) * 32 + kk] = b[i];
            }
        }
        __syncthreads();
        v8bf alo = *(const v8bf*)(Xrow + k0 + g * 8);
        v8bf ahi = *(const v8bf*)(Xrow + k0 + 16 + g * 8);
        v16bf afrag = join16(alo, ahi);
#pragma unroll
        for (int t = 0; t < 4; ++t) {
            const __bf16* wp = &ldsW[(t * 16 + nl) * 32];
            v8bf blo = *(const v8bf*)(wp + g * 8);
            v8bf bhi = *(const v8bf*)(wp + 16 + g * 8);
            v16bf bfrag = join16(blo, bhi);
            acc[t] = __builtin_amdgcn_wmma_f32_16x16x32_bf16(
                false, afrag, false, bfrag, (short)0, acc[t], false, false);
        }
        __syncthreads();
    }

    if (mtile < MTILES) {
#pragma unroll
        for (int t = 0; t < 4; ++t) {
            const int c  = n0 + t * 16 + nl;
            const float bv = bias[c];
#pragma unroll
            for (int r = 0; r < 8; ++r) {
                const int orow = mtile * 16 + r + 8 * g;
                out[(size_t)orow * C + c] = acc[t][r] + bv;
            }
        }
    }
}

// ---------------------------------------------------------------- launch
extern "C" void kernel_launch(void* const* d_in, const int* in_sizes, int n_in,
                              void* d_out, int out_size, void* d_ws, size_t ws_size,
                              hipStream_t stream) {
    const float* x     = (const float*)d_in[0];
    const float* rope  = (const float*)d_in[1];
    const float* Wqkv  = (const float*)d_in[2];
    const float* bqkv  = (const float*)d_in[3];
    const float* Wproj = (const float*)d_in[4];
    const float* bproj = (const float*)d_in[5];
    const float* gamma = (const float*)d_in[6];
    const float* beta  = (const float*)d_in[7];
    float* out = (float*)d_out;

    char* ws = (char*)d_ws;
    size_t off = 0;
    auto alloc = [&](size_t bytes) -> void* {
        void* p = ws + off;
        off = (off + bytes + 255) & ~(size_t)255;
        return p;
    };
    const size_t NBL  = (size_t)M_ROWS * C;     // 8,404,992 (== B*H*L*D)
    __bf16* xb     = (__bf16*)alloc(NBL * 2);
    __bf16* wqkvb  = (__bf16*)alloc((size_t)C * NQKV * 2);
    __bf16* wprojb = (__bf16*)alloc((size_t)C * C * 2);
    __bf16* qb     = (__bf16*)alloc(NBL * 2);
    __bf16* kb     = (__bf16*)alloc(NBL * 2);
    __bf16* vb     = (__bf16*)alloc(NBL * 2);
    float*  attn   = (float*)alloc(NBL * 4);
    __bf16* lnb    = (__bf16*)alloc(NBL * 2);

    // 1. conversions
    {
        int n = (int)NBL;
        MHSA_cvt_bf16<<<(n + 255) / 256, 256, 0, stream>>>(x, xb, n);
        n = C * NQKV;
        MHSA_cvt_bf16<<<(n + 255) / 256, 256, 0, stream>>>(Wqkv, wqkvb, n);
        n = C * C;
        MHSA_cvt_bf16<<<(n + 255) / 256, 256, 0, stream>>>(Wproj, wprojb, n);
    }
    // 2. QKV GEMM + scatter
    MHSA_gemm_qkv<<<dim3((MTILES + 3) / 4, NQKV / 64), 128, 0, stream>>>(
        xb, wqkvb, bqkv, qb, kb, vb);
    // 3. RoPE on q,k
    {
        const int total = B * H * LSP * 32;
        MHSA_rope<<<dim3((total + 255) / 256, 2), 256, 0, stream>>>(qb, kb, rope);
    }
    // 4. flash attention (TDM-staged V)
    MHSA_attn<<<dim3((QTILES + 3) / 4, B * H), 128, 0, stream>>>(qb, kb, vb, attn);
    // 5. LayerNorm
    MHSA_ln<<<M_ROWS, 256, 0, stream>>>(attn, gamma, beta, lnb);
    // 6. projection GEMM
    MHSA_gemm_proj<<<dim3((MTILES + 3) / 4, C / 64), 128, 0, stream>>>(
        lnb, wprojb, bproj, out);
}
